// VQVAE_74775380623865
// MI455X (gfx1250) — compile-verified, matrix operand order checked by reading, not run on hardware
//
#include <hip/hip_runtime.h>
#include <hip/hip_bf16.h>

typedef __bf16 bf16_t;
typedef __attribute__((ext_vector_type(16))) __bf16 v16bf;
typedef __attribute__((ext_vector_type(8)))  float  v8f;

#define B_SZ   8192
#define IN_DIM 1024
#define HID    4096
#define LAT    256
#define KCB    16384
#define LDP    40            // LDS row pitch in bf16 (padded from 32 to spread banks)

union FragAB { v16bf v; uint4 u[2]; };
union FragC  { v8f   v; float f[8]; };

// Async global->LDS 16B copy (gfx1250): no VGPR staging, tracked by ASYNCcnt.
// %0 = LDS byte address (low 32 bits of generic LDS pointer), %1 = global addr.
#define ASYNC_COPY_B128(ldsptr, gptr)                                   \
  asm volatile("global_load_async_to_lds_b128 %0, %1, off"              \
               :: "v"((unsigned)(uintptr_t)(ldsptr)), "v"(gptr) : "memory")
#define WAIT_ASYNC0() asm volatile("s_wait_asynccnt 0x0" ::: "memory")

// ---------------------------------------------------------------- helpers

__global__ void k_f32_to_bf16(const float* __restrict__ src, bf16_t* __restrict__ dst, int n){
  int i = blockIdx.x * blockDim.x + threadIdx.x;
  if (i < n) dst[i] = (bf16_t)src[i];
}

// W[K][N] f32  ->  Wt[N][K] bf16   (K,N multiples of 32)
__global__ void k_transpose_bf16(const float* __restrict__ W, bf16_t* __restrict__ Wt, int K, int N){
  __shared__ float tile[32][33];
  int n = blockIdx.x*32 + threadIdx.x;
  int k = blockIdx.y*32 + threadIdx.y;
  #pragma unroll
  for (int i=0;i<32;i+=8) tile[threadIdx.y+i][threadIdx.x] = W[(size_t)(k+i)*N + n];
  __syncthreads();
  int n2 = blockIdx.x*32 + threadIdx.y;
  int k2 = blockIdx.y*32 + threadIdx.x;
  #pragma unroll
  for (int i=0;i<32;i+=8) Wt[(size_t)(n2+i)*K + k2] = (bf16_t)tile[threadIdx.x][threadIdx.y+i];
}

// sum of squares of a 256-wide row
__global__ void k_rownorm256(const float* __restrict__ X, float* __restrict__ out){
  __shared__ float red[256];
  int row = blockIdx.x, t = threadIdx.x;
  float v = X[(size_t)row*LAT + t];
  red[t] = v*v; __syncthreads();
  for (int s=128; s>0; s>>=1){ if (t<s) red[t]+=red[t+s]; __syncthreads(); }
  if (t==0) out[row] = red[0];
}

__global__ void k_init(unsigned long long* __restrict__ dmin,
                       unsigned int* __restrict__ hist,
                       float* __restrict__ accum){
  int i = blockIdx.x*blockDim.x + threadIdx.x;
  if (i < KCB)  hist[i] = 0u;
  if (i < B_SZ) dmin[i] = ~0ull;
  if (i < 8)    accum[i] = 0.0f;
}

__device__ inline unsigned long long u64_shfl_xor(unsigned long long v, int m){
  unsigned int lo = (unsigned int)v, hi = (unsigned int)(v >> 32);
  lo = (unsigned int)__shfl_xor((int)lo, m, 32);
  hi = (unsigned int)__shfl_xor((int)hi, m, 32);
  return ((unsigned long long)hi << 32) | lo;
}

// ---------------------------------------------------------------- WMMA GEMM
// D[M,N] = act(A[M,K] * Bt[N,K]^T + bias).  128x128 tile, 8 waves (4x2),
// each wave: 2x4 16x16 tiles, K step 32, double-buffered LDS with async copies.

enum { ACT_NONE=0, ACT_RELU=1, ACT_TANH=2 };

template<int ACT>
__global__ __launch_bounds__(256) void k_gemm_bf16(
    const bf16_t* __restrict__ A, const bf16_t* __restrict__ Bt,
    const float* __restrict__ bias,
    bf16_t* __restrict__ Obf, float* __restrict__ Of,
    int M, int N, int K)
{
  __shared__ __align__(16) bf16_t sA[2][128*LDP];
  __shared__ __align__(16) bf16_t sB[2][128*LDP];
  const int tid  = threadIdx.x;
  const int lane = tid & 31;
  const int wv   = tid >> 5;
  const int wr   = wv >> 1, wc = wv & 1;
  const int m0   = blockIdx.y * 128, n0 = blockIdx.x * 128;
  const int lr   = tid >> 2, lc = (tid & 3) << 3;

  FragC acc[2][4];
  #pragma unroll
  for (int i=0;i<2;i++)
    #pragma unroll
    for (int j=0;j<4;j++)
      #pragma unroll
      for (int r=0;r<8;r++) acc[i][j].f[r] = 0.f;

  auto stage = [&](int buf, int kk){
    #pragma unroll
    for (int i=0;i<2;i++){
      int r = lr + i*64;
      ASYNC_COPY_B128(&sA[buf][r*LDP + lc], &A [(size_t)(m0+r)*K + kk + lc]);
      ASYNC_COPY_B128(&sB[buf][r*LDP + lc], &Bt[(size_t)(n0+r)*K + kk + lc]);
    }
  };

  stage(0, 0);
  WAIT_ASYNC0();
  __syncthreads();
  const int ksteps = K >> 5;
  const int krow = (lane >> 4) << 3;   // K sub-chunk per lane half (WMMA A/B layout)
  const int mrow = lane & 15;
  int buf = 0;
  for (int ks=0; ks<ksteps; ks++){
    if (ks+1 < ksteps) stage(buf^1, (ks+1) << 5);
    FragAB a[2], b[4];
    #pragma unroll
    for (int i=0;i<2;i++){
      const bf16_t* p = &sA[buf][(wr*32 + i*16 + mrow)*LDP + krow];
      a[i].u[0] = *(const uint4*)p;
      a[i].u[1] = *(const uint4*)(p + 16);
    }
    #pragma unroll
    for (int j=0;j<4;j++){
      const bf16_t* p = &sB[buf][(wc*64 + j*16 + mrow)*LDP + krow];
      b[j].u[0] = *(const uint4*)p;
      b[j].u[1] = *(const uint4*)(p + 16);
    }
    #pragma unroll
    for (int i=0;i<2;i++)
      #pragma unroll
      for (int j=0;j<4;j++)
        acc[i][j].v = __builtin_amdgcn_wmma_f32_16x16x32_bf16(
            false, a[i].v, false, b[j].v, (short)0, acc[i][j].v, false, false);
    WAIT_ASYNC0();
    __syncthreads();
    buf ^= 1;
  }

  const int col  = lane & 15;
  const int roff = (lane >> 4) << 3;
  #pragma unroll
  for (int j=0;j<4;j++){
    int n = n0 + wc*64 + j*16 + col;
    float bv = bias ? bias[n] : 0.f;
    #pragma unroll
    for (int i=0;i<2;i++){
      int mb = m0 + wr*32 + i*16 + roff;
      #pragma unroll
      for (int r=0;r<8;r++){
        float v = acc[i][j].f[r] + bv;
        if (ACT == ACT_RELU) v = fmaxf(v, 0.f);
        if (ACT == ACT_TANH) v = tanhf(v);
        size_t o = (size_t)(mb + r)*N + n;
        if (Obf) Obf[o] = (bf16_t)v;
        if (Of)  Of[o]  = v;
      }
    }
  }
}

// VQ distance GEMM: acc = z . e_n ; d = ||z||^2 + ||e||^2 - 2 acc ; argmin via
// packed (bits(d)<<32 | n) u64 atomic min (ties -> lowest index, like jnp.argmin).
__global__ __launch_bounds__(256) void k_gemm_vq(
    const bf16_t* __restrict__ A, const bf16_t* __restrict__ Bt,
    const float* __restrict__ znorm, const float* __restrict__ cnorm,
    unsigned long long* __restrict__ dmin, int M, int N, int K)
{
  __shared__ __align__(16) bf16_t sA[2][128*LDP];
  __shared__ __align__(16) bf16_t sB[2][128*LDP];
  const int tid  = threadIdx.x;
  const int lane = tid & 31;
  const int wv   = tid >> 5;
  const int wr   = wv >> 1, wc = wv & 1;
  const int m0   = blockIdx.y * 128, n0 = blockIdx.x * 128;
  const int lr   = tid >> 2, lc = (tid & 3) << 3;

  FragC acc[2][4];
  #pragma unroll
  for (int i=0;i<2;i++)
    #pragma unroll
    for (int j=0;j<4;j++)
      #pragma unroll
      for (int r=0;r<8;r++) acc[i][j].f[r] = 0.f;

  auto stage = [&](int buf, int kk){
    #pragma unroll
    for (int i=0;i<2;i++){
      int r = lr + i*64;
      ASYNC_COPY_B128(&sA[buf][r*LDP + lc], &A [(size_t)(m0+r)*K + kk + lc]);
      ASYNC_COPY_B128(&sB[buf][r*LDP + lc], &Bt[(size_t)(n0+r)*K + kk + lc]);
    }
  };

  stage(0, 0);
  WAIT_ASYNC0();
  __syncthreads();
  const int ksteps = K >> 5;
  const int krow = (lane >> 4) << 3;
  const int mrow = lane & 15;
  int buf = 0;
  for (int ks=0; ks<ksteps; ks++){
    if (ks+1 < ksteps) stage(buf^1, (ks+1) << 5);
    FragAB a[2], b[4];
    #pragma unroll
    for (int i=0;i<2;i++){
      const bf16_t* p = &sA[buf][(wr*32 + i*16 + mrow)*LDP + krow];
      a[i].u[0] = *(const uint4*)p;
      a[i].u[1] = *(const uint4*)(p + 16);
    }
    #pragma unroll
    for (int j=0;j<4;j++){
      const bf16_t* p = &sB[buf][(wc*64 + j*16 + mrow)*LDP + krow];
      b[j].u[0] = *(const uint4*)p;
      b[j].u[1] = *(const uint4*)(p + 16);
    }
    #pragma unroll
    for (int i=0;i<2;i++)
      #pragma unroll
      for (int j=0;j<4;j++)
        acc[i][j].v = __builtin_amdgcn_wmma_f32_16x16x32_bf16(
            false, a[i].v, false, b[j].v, (short)0, acc[i][j].v, false, false);
    WAIT_ASYNC0();
    __syncthreads();
    buf ^= 1;
  }

  const int col  = lane & 15;
  const int roff = (lane >> 4) << 3;
  float cn[4]; int nn[4];
  #pragma unroll
  for (int j=0;j<4;j++){ nn[j] = n0 + wc*64 + j*16 + col; cn[j] = cnorm[nn[j]]; }
  #pragma unroll
  for (int i=0;i<2;i++){
    #pragma unroll
    for (int r=0;r<8;r++){
      int m = m0 + wr*32 + i*16 + roff + r;
      float zn = znorm[m];
      unsigned long long best = ~0ull;
      #pragma unroll
      for (int j=0;j<4;j++){
        float d = fmaxf(zn + cn[j] - 2.0f*acc[i][j].f[r], 0.f);  // >=0 -> bit-monotonic
        unsigned long long key =
            ((unsigned long long)__float_as_uint(d) << 32) | (unsigned int)nn[j];
        if (key < best) best = key;
      }
      #pragma unroll
      for (int mk=1; mk<16; mk<<=1){            // lanes sharing this row
        unsigned long long o = u64_shfl_xor(best, mk);
        if (o < best) best = o;
      }
      if (col == 0) atomicMin(dmin + m, best);
    }
  }
}

// ---------------------------------------------------------------- VQ gather + losses

__global__ void k_vq_gather(const unsigned long long* __restrict__ dmin,
                            const float* __restrict__ cb,
                            const float* __restrict__ z,
                            float* __restrict__ qf, bf16_t* __restrict__ qbf,
                            unsigned int* __restrict__ hist,
                            float* __restrict__ tokens,
                            float* __restrict__ acc0){
  __shared__ float red[256];
  int m = blockIdx.x, t = threadIdx.x;
  unsigned long long key = dmin[m];
  int idx = (int)(unsigned int)(key & 0xffffffffull);
  float cv = cb[(size_t)idx*LAT + t];
  size_t o = (size_t)m*LAT + t;
  qf[o] = cv; qbf[o] = (bf16_t)cv;
  float d = cv - z[o];
  red[t] = d*d; __syncthreads();
  for (int s=128; s>0; s>>=1){ if (t<s) red[t]+=red[t+s]; __syncthreads(); }
  if (t==0){
    atomicAdd(acc0, red[0]);
    atomicAdd(&hist[idx], 1u);
    tokens[m] = (float)idx;
  }
}

__global__ void k_mse(const float* __restrict__ a, const float* __restrict__ b,
                      int n, float* __restrict__ acc){
  __shared__ float red[256];
  int t = threadIdx.x;
  float s = 0.f;
  for (int i = blockIdx.x*256 + t; i < n; i += gridDim.x*256){
    float d = a[i]-b[i]; s += d*d;
  }
  red[t]=s; __syncthreads();
  for (int st=128; st>0; st>>=1){ if (t<st) red[t]+=red[t+st]; __syncthreads(); }
  if (t==0) atomicAdd(acc, red[0]);
}

__global__ void k_finalize(const unsigned int* __restrict__ hist,
                           const float* __restrict__ accum,
                           float* __restrict__ out4){
  __shared__ float red[256];
  int t = threadIdx.x;
  float s = 0.f;
  for (int i=t; i<KCB; i+=256){
    float p = (float)hist[i] * (1.0f/(float)B_SZ);
    s += p * logf(p + 1e-10f);
  }
  red[t]=s; __syncthreads();
  for (int st=128; st>0; st>>=1){ if (t<st) red[t]+=red[t+st]; __syncthreads(); }
  if (t==0){
    float vq = 1.25f * accum[0] * (1.0f/((float)B_SZ*(float)LAT));   // (1+COMMIT)*mse(q,z)
    float rl = accum[1] * (1.0f/((float)B_SZ*(float)IN_DIM));
    out4[0]=vq; out4[1]=rl; out4[2]=vq+rl; out4[3]=expf(-red[0]);
  }
}

// ---------------------------------------------------------------- launcher

extern "C" void kernel_launch(void* const* d_in, const int* in_sizes, int n_in,
                              void* d_out, int out_size, void* d_ws, size_t ws_size,
                              hipStream_t stream) {
  (void)in_sizes; (void)n_in; (void)out_size; (void)ws_size;

  const float* x    = (const float*)d_in[0];
  const float* eW1  = (const float*)d_in[1];
  const float* eb1  = (const float*)d_in[2];
  const float* eW2  = (const float*)d_in[3];
  const float* eb2  = (const float*)d_in[4];
  const float* eW3  = (const float*)d_in[5];
  const float* eb3  = (const float*)d_in[6];
  const float* cb   = (const float*)d_in[7];
  const float* dW1  = (const float*)d_in[8];
  const float* db1  = (const float*)d_in[9];
  const float* dW2  = (const float*)d_in[10];
  const float* db2  = (const float*)d_in[11];
  const float* dW3  = (const float*)d_in[12];
  const float* db3  = (const float*)d_in[13];

  float* recon  = (float*)d_out;                       // [B, IN_DIM]
  float* tokens = recon + (size_t)B_SZ*IN_DIM;         // [B]
  float* scal   = tokens + B_SZ;                       // vq, recon, total, perp

  size_t off = 0; char* ws = (char*)d_ws;
  auto take = [&](size_t bytes)->void*{
    void* p = (void*)(ws + off);
    off += (bytes + 255) & ~(size_t)255;
    return p;
  };
  bf16_t* xbf   = (bf16_t*)take((size_t)B_SZ*IN_DIM*2);
  bf16_t* h1bf  = (bf16_t*)take((size_t)B_SZ*HID*2);
  bf16_t* h2bf  = (bf16_t*)take((size_t)B_SZ*HID*2);
  float*  zf    = (float*) take((size_t)B_SZ*LAT*4);
  bf16_t* zbf   = (bf16_t*)take((size_t)B_SZ*LAT*2);
  float*  qf    = (float*) take((size_t)B_SZ*LAT*4);
  bf16_t* qbf   = (bf16_t*)take((size_t)B_SZ*LAT*2);
  bf16_t* eW1t  = (bf16_t*)take((size_t)IN_DIM*HID*2);
  bf16_t* eW2t  = (bf16_t*)take((size_t)HID*HID*2);
  bf16_t* eW3t  = (bf16_t*)take((size_t)HID*LAT*2);
  bf16_t* dW1t  = (bf16_t*)take((size_t)LAT*HID*2);
  bf16_t* dW2t  = (bf16_t*)take((size_t)HID*HID*2);
  bf16_t* dW3t  = (bf16_t*)take((size_t)HID*IN_DIM*2);
  bf16_t* cbbf  = (bf16_t*)take((size_t)KCB*LAT*2);
  float*  cnorm = (float*) take((size_t)KCB*4);
  float*  znorm = (float*) take((size_t)B_SZ*4);
  unsigned long long* dmin = (unsigned long long*)take((size_t)B_SZ*8);
  unsigned int* hist = (unsigned int*)take((size_t)KCB*4);
  float* accum = (float*)take(8*4);

  dim3 tb(32, 8);
  // precision conversions / weight transposes (Wt[N][K] bf16)
  k_f32_to_bf16<<<(B_SZ*IN_DIM)/256, 256, 0, stream>>>(x, xbf, B_SZ*IN_DIM);
  k_f32_to_bf16<<<(KCB*LAT)/256,    256, 0, stream>>>(cb, cbbf, KCB*LAT);   // already [N,K]
  k_transpose_bf16<<<dim3(HID/32,    IN_DIM/32), tb, 0, stream>>>(eW1, eW1t, IN_DIM, HID);
  k_transpose_bf16<<<dim3(HID/32,    HID/32),    tb, 0, stream>>>(eW2, eW2t, HID,    HID);
  k_transpose_bf16<<<dim3(LAT/32,    HID/32),    tb, 0, stream>>>(eW3, eW3t, HID,    LAT);
  k_transpose_bf16<<<dim3(HID/32,    LAT/32),    tb, 0, stream>>>(dW1, dW1t, LAT,    HID);
  k_transpose_bf16<<<dim3(HID/32,    HID/32),    tb, 0, stream>>>(dW2, dW2t, HID,    HID);
  k_transpose_bf16<<<dim3(IN_DIM/32, HID/32),    tb, 0, stream>>>(dW3, dW3t, HID,    IN_DIM);
  k_rownorm256<<<KCB, 256, 0, stream>>>(cb, cnorm);
  k_init<<<KCB/256, 256, 0, stream>>>(dmin, hist, accum);

  // encoder
  k_gemm_bf16<ACT_RELU><<<dim3(HID/128, B_SZ/128), 256, 0, stream>>>(
      xbf,  eW1t, eb1, h1bf, nullptr, B_SZ, HID, IN_DIM);
  k_gemm_bf16<ACT_RELU><<<dim3(HID/128, B_SZ/128), 256, 0, stream>>>(
      h1bf, eW2t, eb2, h2bf, nullptr, B_SZ, HID, HID);
  k_gemm_bf16<ACT_TANH><<<dim3(LAT/128, B_SZ/128), 256, 0, stream>>>(
      h2bf, eW3t, eb3, zbf, zf, B_SZ, LAT, HID);

  // vector quantization
  k_rownorm256<<<B_SZ, 256, 0, stream>>>(zf, znorm);
  k_gemm_vq<<<dim3(KCB/128, B_SZ/128), 256, 0, stream>>>(
      zbf, cbbf, znorm, cnorm, dmin, B_SZ, KCB, LAT);
  k_vq_gather<<<B_SZ, 256, 0, stream>>>(dmin, cb, zf, qf, qbf, hist, tokens, accum + 0);

  // decoder (straight-through forward value: q)
  k_gemm_bf16<ACT_RELU><<<dim3(HID/128, B_SZ/128), 256, 0, stream>>>(
      qbf,  dW1t, db1, h1bf, nullptr, B_SZ, HID, LAT);
  k_gemm_bf16<ACT_RELU><<<dim3(HID/128, B_SZ/128), 256, 0, stream>>>(
      h1bf, dW2t, db2, h2bf, nullptr, B_SZ, HID, HID);
  k_gemm_bf16<ACT_NONE><<<dim3(IN_DIM/128, B_SZ/128), 256, 0, stream>>>(
      h2bf, dW3t, db3, nullptr, recon, B_SZ, IN_DIM, HID);

  // losses + perplexity
  k_mse<<<1024, 256, 0, stream>>>(recon, x, B_SZ*IN_DIM, accum + 1);
  k_finalize<<<1, 256, 0, stream>>>(hist, accum, scal);
}
